// LinformerSelfAttention_9878424780892
// MI455X (gfx1250) — matrix operand order repaired and checked by reading, compiled, not run on hardware
//
#include <hip/hip_runtime.h>

// ---------------------------------------------------------------------------
// Linformer self-attention, fully fused per batch element, CDNA5 WMMA (f16).
// B=4096, N=68, D=128, H=8, DH=16, L=50.
// ---------------------------------------------------------------------------

#define BATCH  4096
#define SEQ    68
#define DMODEL 128
#define NHEAD  8
#define DHEAD  16
#define LRANK  50
#define NPAD   96   // seq padded to multiple of 32 (WMMA K dim)
#define MPAD   80   // seq padded to multiple of 16 (WMMA M dim)
#define LPAD   64   // rank padded to multiple of 32

typedef _Float16 half4 __attribute__((ext_vector_type(4)));
typedef _Float16 half8 __attribute__((ext_vector_type(8)));
typedef _Float16 v16h  __attribute__((ext_vector_type(16)));
typedef float    v8f   __attribute__((ext_vector_type(8)));

// f16 workspace layout (element offsets)
#define WS_WQ 0
#define WS_WK 16384
#define WS_WV 32768
#define WS_WO 49152
#define WS_ET 65536            // E^T padded [LPAD][NPAD]
#define WS_FT 71680            // F^T padded [LPAD][NPAD]
#define WS_TOTAL 77824

// LDS layout (half elements). Total 118784 halves = 237,568 B (< 320 KB/WGP).
#define SM_XS 0                // x     [96][128] f16
#define SM_QS 12288            // Q     [96][128]
#define SM_KS 24576            // K     [96][128]
#define SM_VS 36864            // V     [96][128]
#define SM_OS 49152            // head outputs [96][128]
#define SM_KP 61440            // Kp    [8][64][16]
#define SM_VP 69632            // Vp    [8][64][16]
#define SM_SP 77824            // scores/probs [8][80][64]
#define SM_TOTAL 118784

__device__ __forceinline__ v16h pack16(half8 lo, half8 hi) {
  v16h r;
#pragma unroll
  for (int i = 0; i < 8; ++i) { r[i] = lo[i]; r[i + 8] = hi[i]; }
  return r;
}

// A fragment (16x32, f16), row-major source. Lanes 0-15 hold
// row M=lane with K = 0..15; lanes 16-31 hold M=lane-16 with
// K = 16..31. One contiguous 32B run per lane.
__device__ __forceinline__ v16h load_a(const _Float16* base, int rowStride,
                                       int row, int kbase, int lane) {
  const int klo = (lane < 16) ? 0 : 16;
  const _Float16* p = base + row * rowStride + kbase + klo;
  half8 lo = *(const half8*)p;
  half8 hi = *(const half8*)(p + 8);
  return pack16(lo, hi);
}

// A fragment with only K=0..15 valid (upper 16 K zero-padded).
__device__ __forceinline__ v16h load_a_k16(const _Float16* base, int rowStride,
                                           int row, int kbase, int lane) {
  const _Float16* p = base + row * rowStride + kbase;
  half8 lo = *(const half8*)p;
  half8 hi = *(const half8*)(p + 8);
  const bool act = lane < 16;
  v16h r;
#pragma unroll
  for (int i = 0; i < 8; ++i) { r[i] = act ? lo[i] : (_Float16)0; r[i + 8] = act ? hi[i] : (_Float16)0; }
  return r;
}

// B fragment (32x16, f16) from a [col][k] row-major panel (e.g. torch weight
// [d_out][d_in]). Lanes 0-15: col=lane, K=0..15; lanes 16-31: col=lane-16,
// K=16..31 — one contiguous 32B run per lane.
__device__ __forceinline__ v16h load_b_contig(const _Float16* base, int rowStride,
                                              int col, int kbase, int lane) {
  const int klo = (lane < 16) ? 0 : 16;
  const _Float16* p = base + col * rowStride + kbase + klo;
  half8 lo = *(const half8*)p;
  half8 hi = *(const half8*)(p + 8);
  return pack16(lo, hi);
}

// B fragment with only K=0..15 valid (upper K zero); source [col][k<=16].
__device__ __forceinline__ v16h load_b_k16(const _Float16* base, int rowStride,
                                           int col, int lane) {
  const _Float16* p = base + col * rowStride;
  half8 lo = *(const half8*)p;
  half8 hi = *(const half8*)(p + 8);
  const bool act = lane < 16;
  v16h r;
#pragma unroll
  for (int i = 0; i < 8; ++i) {
    r[i]     = act ? lo[i] : (_Float16)0;
    r[i + 8] = act ? hi[i] : (_Float16)0;
  }
  return r;
}

// B fragment from a [k][col] row-major panel (strided gather; small panels,
// reused across all M tiles so the cost is amortized).
__device__ __forceinline__ v16h load_b_strided(const _Float16* base, int rowStride,
                                               int col, int kbase, int lane) {
  const int klo = (lane < 16) ? 0 : 16;
  v16h r;
#pragma unroll
  for (int t = 0; t < 16; ++t)
    r[t] = base[(kbase + klo + t) * rowStride + col];
  return r;
}

// Store a 16x16 f32 accumulator as f16: VGPR r holds (M = r (+8 for hi lanes),
// N = lane&15).
__device__ __forceinline__ void store_d_f16(_Float16* base, int rowStride,
                                            int row0, int col0, int lane,
                                            v8f acc, float scale) {
  const int col = col0 + (lane & 15);
  const int rb  = row0 + ((lane >= 16) ? 8 : 0);
#pragma unroll
  for (int r = 0; r < 8; ++r)
    base[(rb + r) * rowStride + col] = (_Float16)(acc[r] * scale);
}

__device__ __forceinline__ v8f wmma16(v16h a, v16h b, v8f c) {
  return __builtin_amdgcn_wmma_f32_16x16x32_f16(false, a, false, b,
                                                (short)0, c, false, false);
}

// ---------------------------------------------------------------------------
// Prep: fp32 weights -> f16; build zero-padded E^T / F^T panels in workspace.
// ---------------------------------------------------------------------------
__global__ __launch_bounds__(256)
void linformer_prep(const float* __restrict__ Wq, const float* __restrict__ Wk,
                    const float* __restrict__ Wv, const float* __restrict__ Wo,
                    const float* __restrict__ E,  const float* __restrict__ F,
                    _Float16* __restrict__ ws16) {
  const int idx = blockIdx.x * 256 + threadIdx.x;
  if (idx < 16384) {
    ws16[WS_WQ + idx] = (_Float16)Wq[idx];
  } else if (idx < 32768) {
    ws16[WS_WK + (idx - 16384)] = (_Float16)Wk[idx - 16384];
  } else if (idx < 49152) {
    ws16[WS_WV + (idx - 32768)] = (_Float16)Wv[idx - 32768];
  } else if (idx < 65536) {
    ws16[WS_WO + (idx - 49152)] = (_Float16)Wo[idx - 49152];
  } else if (idx < 71680) {
    const int local = idx - 65536;
    const int l = local / NPAD, n = local % NPAD;
    ws16[WS_ET + local] =
        (l < LRANK && n < SEQ) ? (_Float16)E[n * LRANK + l] : (_Float16)0;
  } else if (idx < WS_TOTAL) {
    const int local = idx - 71680;
    const int l = local / NPAD, n = local % NPAD;
    ws16[WS_FT + local] =
        (l < LRANK && n < SEQ) ? (_Float16)F[n * LRANK + l] : (_Float16)0;
  }
}

// ---------------------------------------------------------------------------
// Fused kernel: one workgroup (8 wave32) per batch element; one wave per head.
// ---------------------------------------------------------------------------
__global__ __launch_bounds__(256)
void linformer_fused(const float* __restrict__ x,
                     const _Float16* __restrict__ ws16,
                     const float* __restrict__ bo,
                     float* __restrict__ out) {
  extern __shared__ _Float16 smem[];
  _Float16* xs = smem + SM_XS;
  _Float16* qs = smem + SM_QS;
  _Float16* ks = smem + SM_KS;
  _Float16* vs = smem + SM_VS;
  _Float16* os = smem + SM_OS;
  _Float16* kp = smem + SM_KP;
  _Float16* vp = smem + SM_VP;
  _Float16* sp = smem + SM_SP;

  const int tid  = threadIdx.x;
  const int lane = tid & 31;
  const int wave = tid >> 5;
  const int b    = blockIdx.x;

  const _Float16* Wq16 = ws16 + WS_WQ;
  const _Float16* Wk16 = ws16 + WS_WK;
  const _Float16* Wv16 = ws16 + WS_WV;
  const _Float16* Wo16 = ws16 + WS_WO;
  const _Float16* Et   = ws16 + WS_ET;
  const _Float16* Ft   = ws16 + WS_FT;

  // Warm the shared weight panels toward cache (global_prefetch_b8).
  for (int i = tid; i < 512; i += 256)
    __builtin_prefetch(ws16 + i * 128, 0, 1);

  // ---- Phase A: stage x[b] into LDS as f16, zero the padding -------------
  const float* xb = x + (size_t)b * (SEQ * DMODEL);
  for (int i4 = tid; i4 < (SEQ * DMODEL) / 4; i4 += 256) {
    const float4 v = ((const float4*)xb)[i4];
    half4 h; h[0] = (_Float16)v.x; h[1] = (_Float16)v.y;
             h[2] = (_Float16)v.z; h[3] = (_Float16)v.w;
    ((half4*)xs)[i4] = h;
  }
  for (int i = SEQ * DMODEL + tid; i < NPAD * DMODEL; i += 256) xs[i] = (_Float16)0;
  for (int i = MPAD * DMODEL + tid; i < NPAD * DMODEL; i += 256) {
    ks[i] = (_Float16)0;  // K/V rows 80..95 feed the padded K-dim of E^T/F^T
    vs[i] = (_Float16)0;
  }
  __syncthreads();

  // ---- Phase B: Q = x Wq^T, K = x Wk^T, V = x Wv^T (f16 in LDS) ----------
  // 3 mats x 5 Mtiles x 8 Ntiles = 120 tiles over 8 waves, 4 k-steps each.
  for (int t = wave; t < 120; t += 8) {
    const int mat   = t / 40;
    const int rem   = t % 40;
    const int mtile = rem >> 3;
    const int ntile = rem & 7;
    const _Float16* W   = (mat == 0) ? Wq16 : (mat == 1) ? Wk16 : Wv16;
    _Float16*       dst = (mat == 0) ? qs   : (mat == 1) ? ks   : vs;
    v8f acc = {};
#pragma unroll
    for (int kstep = 0; kstep < 4; ++kstep) {
      v16h a  = load_a(xs, DMODEL, mtile * 16 + (lane & 15), kstep * 32, lane);
      v16h bb = load_b_contig(W, DMODEL, ntile * 16 + (lane & 15), kstep * 32, lane);
      acc = wmma16(a, bb, acc);
    }
    store_d_f16(dst, DMODEL, mtile * 16, ntile * 16, lane, acc, 1.0f);
  }
  __syncthreads();

  // ---- Phase C: Kp = E^T K_head, Vp = F^T V_head (per head = per wave) ---
  {
    const int h = wave;
    _Float16* kph = kp + h * (LPAD * DHEAD);
    _Float16* vph = vp + h * (LPAD * DHEAD);

    v16h kb0 = load_b_strided(ks, DMODEL, h * 16 + (lane & 15), 0,  lane);
    v16h kb1 = load_b_strided(ks, DMODEL, h * 16 + (lane & 15), 32, lane);
    v16h kb2 = load_b_strided(ks, DMODEL, h * 16 + (lane & 15), 64, lane);
#pragma unroll
    for (int mt = 0; mt < 4; ++mt) {
      v8f acc = {};
      acc = wmma16(load_a(Et, NPAD, mt * 16 + (lane & 15), 0,  lane), kb0, acc);
      acc = wmma16(load_a(Et, NPAD, mt * 16 + (lane & 15), 32, lane), kb1, acc);
      acc = wmma16(load_a(Et, NPAD, mt * 16 + (lane & 15), 64, lane), kb2, acc);
      store_d_f16(kph, DHEAD, mt * 16, 0, lane, acc, 1.0f);
    }

    v16h vb0 = load_b_strided(vs, DMODEL, h * 16 + (lane & 15), 0,  lane);
    v16h vb1 = load_b_strided(vs, DMODEL, h * 16 + (lane & 15), 32, lane);
    v16h vb2 = load_b_strided(vs, DMODEL, h * 16 + (lane & 15), 64, lane);
#pragma unroll
    for (int mt = 0; mt < 4; ++mt) {
      v8f acc = {};
      acc = wmma16(load_a(Ft, NPAD, mt * 16 + (lane & 15), 0,  lane), vb0, acc);
      acc = wmma16(load_a(Ft, NPAD, mt * 16 + (lane & 15), 32, lane), vb1, acc);
      acc = wmma16(load_a(Ft, NPAD, mt * 16 + (lane & 15), 64, lane), vb2, acc);
      store_d_f16(vph, DHEAD, mt * 16, 0, lane, acc, 1.0f);
    }
  }
  __syncthreads();

  // ---- Phase D: scores = (Q_head Kp^T) * DH^-0.5 -------------------------
  {
    const int h = wave;
    _Float16*       sph = sp + h * (MPAD * LPAD);
    const _Float16* kph = kp + h * (LPAD * DHEAD);
#pragma unroll
    for (int mt = 0; mt < 5; ++mt) {
      v16h a = load_a_k16(qs, DMODEL, mt * 16 + (lane & 15), h * 16, lane);
#pragma unroll
      for (int nt = 0; nt < 4; ++nt) {
        v16h bb = load_b_k16(kph, DHEAD, nt * 16 + (lane & 15), lane);
        v8f acc = {};
        acc = wmma16(a, bb, acc);
        store_d_f16(sph, LPAD, mt * 16, nt * 16, lane, acc, 0.25f);  // 1/sqrt(16)
      }
    }
  }
  __syncthreads();

  // ---- Softmax over L=50 (per head, f32 math, f16 storage) ---------------
  {
    const int h = wave;
    _Float16* sph = sp + h * (MPAD * LPAD);
    for (int r = lane; r < SEQ; r += 32) {
      _Float16* row = sph + r * LPAD;
      float m = -1e30f;
      for (int l = 0; l < LRANK; ++l) m = fmaxf(m, (float)row[l]);
      float s = 0.f;
      for (int l = 0; l < LRANK; ++l) s += __expf((float)row[l] - m);
      const float inv = 1.f / s;
      for (int l = 0; l < LRANK; ++l)
        row[l] = (_Float16)(__expf((float)row[l] - m) * inv);
      for (int l = LRANK; l < LPAD; ++l) row[l] = (_Float16)0;
    }
  }
  __syncthreads();

  // ---- Phase E: out_head = probs @ Vp ------------------------------------
  {
    const int h = wave;
    _Float16*       sph = sp + h * (MPAD * LPAD);
    const _Float16* vph = vp + h * (LPAD * DHEAD);
    v16h vb0 = load_b_strided(vph, DHEAD, lane & 15, 0,  lane);
    v16h vb1 = load_b_strided(vph, DHEAD, lane & 15, 32, lane);
#pragma unroll
    for (int mt = 0; mt < 5; ++mt) {
      v8f acc = {};
      acc = wmma16(load_a(sph, LPAD, mt * 16 + (lane & 15), 0,  lane), vb0, acc);
      acc = wmma16(load_a(sph, LPAD, mt * 16 + (lane & 15), 32, lane), vb1, acc);
      store_d_f16(os, DMODEL, mt * 16, h * 16, lane, acc, 1.0f);
    }
  }
  __syncthreads();

  // ---- Phase F: y = concat(out_heads) @ Wo^T + bo (fp32 to HBM) ----------
  {
    float* outb = out + (size_t)b * (SEQ * DMODEL);
    for (int t = wave; t < 40; t += 8) {
      const int mtile = t >> 3;
      const int ntile = t & 7;
      v8f acc = {};
#pragma unroll
      for (int kstep = 0; kstep < 4; ++kstep) {
        v16h a  = load_a(os, DMODEL, mtile * 16 + (lane & 15), kstep * 32, lane);
        v16h bb = load_b_contig(Wo16, DMODEL, ntile * 16 + (lane & 15), kstep * 32, lane);
        acc = wmma16(a, bb, acc);
      }
      const int col  = ntile * 16 + (lane & 15);
      const float bi = bo[col];
      const int rb   = mtile * 16 + ((lane >= 16) ? 8 : 0);
#pragma unroll
      for (int r = 0; r < 8; ++r) {
        const int row = rb + r;
        if (row < SEQ) outb[row * DMODEL + col] = acc[r] + bi;
      }
    }
  }
}

// ---------------------------------------------------------------------------
extern "C" void kernel_launch(void* const* d_in, const int* in_sizes, int n_in,
                              void* d_out, int out_size, void* d_ws, size_t ws_size,
                              hipStream_t stream) {
  (void)in_sizes; (void)n_in; (void)out_size; (void)ws_size;
  const float* x  = (const float*)d_in[0];
  const float* Wq = (const float*)d_in[1];
  const float* Wk = (const float*)d_in[2];
  const float* Wv = (const float*)d_in[3];
  const float* Wo = (const float*)d_in[4];
  const float* bo = (const float*)d_in[5];
  const float* E  = (const float*)d_in[6];
  const float* F  = (const float*)d_in[7];
  _Float16* ws16 = (_Float16*)d_ws;        // needs WS_TOTAL*2 = 155,648 B
  float* out = (float*)d_out;

  linformer_prep<<<(WS_TOTAL + 255) / 256, 256, 0, stream>>>(Wq, Wk, Wv, Wo, E, F, ws16);

  const size_t smem_bytes = (size_t)SM_TOTAL * sizeof(_Float16);  // 237,568 B
  linformer_fused<<<BATCH, 256, smem_bytes, stream>>>(x, ws16, bo, out);
}
